// GraphReasoning_17463337026078
// MI455X (gfx1250) — matrix operand order repaired and checked by reading, not compiled, not run on hardware
//
#include <hip/hip_runtime.h>

// ---------------------------------------------------------------------------
// Problem constants (reference: B=4, N=4096, D=128)
// ---------------------------------------------------------------------------
#define BB 4
#define NN 4096
#define DD 128
#define DV 144          // V padded with 16 all-ones columns (row-sum trick)
#define DVL 160         // V tile LDS stride (uniform async chunk count)
#define KT 64           // keys per tile

typedef __bf16 bf16_t;
typedef __attribute__((ext_vector_type(16))) __bf16    v16bf;
typedef __attribute__((ext_vector_type(8)))  float     v8f;
typedef __attribute__((ext_vector_type(4)))  unsigned  v4u;

union ABu { v4u u[2]; v16bf v; };

// B-fragment (16-bit, 32x16): lane L holds column L%16; p = &M[k0 + 8*(L/16)][n];
// elements 0..7 = K rows p+0..7, elements 8..15 = K rows p+16..23 (stride ld).
__device__ __forceinline__ v16bf load_bfrag16(const bf16_t* p, int ld) {
    v16bf r;
#pragma unroll
    for (int i = 0; i < 8; ++i) {
        r[i]     = p[i * ld];
        r[i + 8] = p[(16 + i) * ld];
    }
    return r;
}

// A-fragment (16-bit, 16x32): lane L holds row L%16; elements 0..7 = K k0..k0+7,
// elements 8..15 = K k0+16..k0+23, with k0 = 32*kt + 8*(L/16).
__device__ __forceinline__ v16bf load_afrag_bf16(const bf16_t* row, int k0) {
    ABu t;
    t.u[0] = *(const v4u*)(row + k0);
    t.u[1] = *(const v4u*)(row + k0 + 16);
    return t.v;
}

__device__ __forceinline__ v16bf load_afrag_f32(const float* row, int k0) {
    v16bf r;
#pragma unroll
    for (int i = 0; i < 8; ++i) {
        r[i]     = (__bf16)row[k0 + i];
        r[i + 8] = (__bf16)row[k0 + 16 + i];
    }
    return r;
}

__device__ __forceinline__ v8f wmma_bf16(v16bf a, v16bf b, v8f c) {
    return __builtin_amdgcn_wmma_f32_16x16x32_bf16(
        false, a, false, b, (short)0, c, false, false);
}

// Reduce across the 16 lanes of each wave half (C-layout columns).
__device__ __forceinline__ float half_reduce_max(float v) {
#pragma unroll
    for (int m = 1; m < 16; m <<= 1) v = fmaxf(v, __shfl_xor(v, m, 32));
    return v;
}

// Wave-local LDS ordering (stores by this wave -> loads by this wave).
__device__ __forceinline__ void wave_lds_fence() {
    asm volatile("s_wait_dscnt 0" ::: "memory");
}

// Async Global -> LDS copy of 16 bytes (tracked by ASYNCcnt).
__device__ __forceinline__ void async_copy_b128(unsigned lds_off, const void* gaddr) {
    asm volatile("global_load_async_to_lds_b128 %0, %1, off"
                 :: "v"(lds_off), "v"(gaddr) : "memory");
}
__device__ __forceinline__ void wait_async0() {
    asm volatile("s_wait_asynccnt 0x0" ::: "memory");
}

// ---------------------------------------------------------------------------
// Kernel 1: q = X@Wq^T + bq (row-major bf16), kT = (X@Wk^T + bk)^T (bf16),
//           V = [bf16(X) | ones(16)]  ([B*N, 144]).
// One block = 128 rows of flattened [B*N, D].
// ---------------------------------------------------------------------------
__global__ __launch_bounds__(256) void qkv_proj_kernel(
    const float* __restrict__ X,
    const float* __restrict__ Wq, const float* __restrict__ bq,
    const float* __restrict__ Wk, const float* __restrict__ bk,
    unsigned short* __restrict__ q_raw,
    unsigned short* __restrict__ kT_raw,
    unsigned short* __restrict__ v_raw)
{
    bf16_t* q_ws  = reinterpret_cast<bf16_t*>(q_raw);   // [B*N, D]
    bf16_t* kT_ws = reinterpret_cast<bf16_t*>(kT_raw);  // [B, D, N]
    bf16_t* v_ws  = reinterpret_cast<bf16_t*>(v_raw);   // [B*N, DV]

    __shared__ __attribute__((aligned(16))) bf16_t wt_q[DD][DD]; // Wq^T
    __shared__ __attribute__((aligned(16))) bf16_t wt_k[DD][DD]; // Wk^T

    const int tid = threadIdx.x;
    const long rowBase = (long)blockIdx.x * 128;

    for (int idx = tid; idx < DD * DD; idx += 256) {
        int e = idx >> 7, d = idx & 127;
        wt_q[d][e] = (__bf16)Wq[idx];
        wt_k[d][e] = (__bf16)Wk[idx];
    }
    // V tile copy with ones-padding (columns 128..143 = 1.0).
    for (int idx = tid; idx < 128 * DV; idx += 256) {
        long g = rowBase + idx / DV;
        int  d = idx % DV;
        v_ws[g * DV + d] = (d < DD) ? (__bf16)X[g * DD + d] : (__bf16)1.0f;
    }
    __syncthreads();

    const int wave = tid >> 5, lane = tid & 31;
    const int half = lane >> 4, l16 = lane & 15;
    const long r0 = rowBase + wave * 16;

    v16bf a[4];
    const float* xrow = X + (r0 + l16) * DD;
#pragma unroll
    for (int kt = 0; kt < 4; ++kt) a[kt] = load_afrag_f32(xrow, 32 * kt + 8 * half);

#pragma unroll
    for (int et = 0; et < 8; ++et) {
        const int e = et * 16 + l16;
        // Preload all B fragments first -> one DS wait, back-to-back WMMAs.
        v16bf bqf[4], bkf[4];
#pragma unroll
        for (int kt = 0; kt < 4; ++kt) {
            const int d = 32 * kt + 8 * half;
            bqf[kt] = load_bfrag16(&wt_q[d][e], DD);
            bkf[kt] = load_bfrag16(&wt_k[d][e], DD);
        }
        v8f accq = {}; v8f acck = {};
#pragma unroll
        for (int kt = 0; kt < 4; ++kt) accq = wmma_bf16(a[kt], bqf[kt], accq);
#pragma unroll
        for (int kt = 0; kt < 4; ++kt) acck = wmma_bf16(a[kt], bkf[kt], acck);

        const float biasq = bq[e], biask = bk[e];
#pragma unroll
        for (int v = 0; v < 8; ++v) {                  // C-layout rows v + 8*half
            const long row = r0 + v + 8 * half;
            const long b   = row >> 12;
            const long n   = row & (NN - 1);
            q_ws[row * DD + e]                 = (__bf16)(accq[v] + biasq);
            kT_ws[(b * DD + e) * (long)NN + n] = (__bf16)(acck[v] + biask);
        }
    }
}

// ---------------------------------------------------------------------------
// Kernel 2: flash attention (64-key tiles, async double-buffered LDS staging)
//           + fused out = relu(O @ Wg^T + bg).
// ---------------------------------------------------------------------------
__global__ __launch_bounds__(256) void attn_out_kernel(
    const unsigned short* __restrict__ q_raw,
    const unsigned short* __restrict__ kT_raw,
    const unsigned short* __restrict__ v_raw,
    const float* __restrict__ Wg, const float* __restrict__ bg,
    float* __restrict__ out)
{
    const bf16_t* q_ws  = reinterpret_cast<const bf16_t*>(q_raw);
    const bf16_t* kT_ws = reinterpret_cast<const bf16_t*>(kT_raw);
    const bf16_t* v_ws  = reinterpret_cast<const bf16_t*>(v_raw);

    __shared__ __attribute__((aligned(16))) bf16_t wt_g[DD][DD];        // 32KB
    __shared__ __attribute__((aligned(16))) bf16_t kTs[2][DD][KT];      // 2x16KB
    __shared__ __attribute__((aligned(16))) bf16_t Vs[2][KT][DVL];      // 2x20KB
    __shared__ __attribute__((aligned(16))) bf16_t p_lds[8][16][KT];    // 16KB
    __shared__ __attribute__((aligned(16))) bf16_t o_lds[8][16][DD];    // 32KB

    const int tid = threadIdx.x;
    const int b     = blockIdx.x >> 5;
    const int qtile = blockIdx.x & 31;
    const long nBase = (long)b * NN;

    const bf16_t* kTb = kT_ws + (long)b * DD * NN;
    const bf16_t* vb  = v_ws + nBase * DV;
    const long vRemain = ((long)BB * NN - nBase) * DV;   // OOB guard for V tile

    for (int idx = tid; idx < DD * DD; idx += 256) {
        int e = idx >> 7, d = idx & 127;
        wt_g[d][e] = (__bf16)Wg[idx];
    }

    // --- async stage of one 64-key tile (kT: 1024 chunks, V: 1280 chunks) ---
    auto stage_tile = [&](int key0, int buf) {
#pragma unroll
        for (int i = 0; i < 4; ++i) {                  // kT: 4 chunks/thread
            int c = tid + i * 256;
            int d = c >> 3, k8 = (c & 7) * 8;
            async_copy_b128((unsigned)(uintptr_t)&kTs[buf][d][k8],
                            kTb + (long)d * NN + key0 + k8);
        }
#pragma unroll
        for (int i = 0; i < 5; ++i) {                  // V: 5 chunks/thread
            int c = tid + i * 256;
            int key = c / 20, d8 = (c % 20) * 8;       // LDS stride DVL=160
            long g = (long)(key0 + key) * DV + d8;
            if (g + 8 <= vRemain)
                async_copy_b128((unsigned)(uintptr_t)&Vs[buf][key][d8], vb + g);
        }
    };

    stage_tile(0, 0);   // prologue: tile 0 in flight

    const int wave = tid >> 5, lane = tid & 31;
    const int half = lane >> 4, l16 = lane & 15;
    const int qr0  = qtile * 128 + wave * 16;

    v16bf aq[4];
    const bf16_t* qrow = q_ws + (nBase + qr0 + l16) * DD;
#pragma unroll
    for (int kt = 0; kt < 4; ++kt) aq[kt] = load_afrag_bf16(qrow, 32 * kt + 8 * half);

    v8f   o[9];                      // o[8] = running row-sum (ones columns)
    float m[8];
#pragma unroll
    for (int dt = 0; dt < 9; ++dt) o[dt] = (v8f){};
#pragma unroll
    for (int v = 0; v < 8; ++v) m[v] = -__builtin_inff();

    for (int it = 0; it < NN / KT; ++it) {
        wait_async0();           // tile `it` landed (this wave's copies)
        __syncthreads();         // all waves landed AND done reading old buffer
        if (it + 1 < NN / KT) stage_tile((it + 1) * KT, (it + 1) & 1);
        const int buf = it & 1;

        // S = Q @ K^T for 64 keys (four 16-wide column tiles).
        v8f s[4];
#pragma unroll
        for (int nt = 0; nt < 4; ++nt) {
            v16bf bf[4];
#pragma unroll
            for (int kc = 0; kc < 4; ++kc)
                bf[kc] = load_bfrag16(&kTs[buf][32 * kc + 8 * half][nt * 16 + l16], KT);
            v8f acc = {};
#pragma unroll
            for (int kc = 0; kc < 4; ++kc) acc = wmma_bf16(aq[kc], bf[kc], acc);
            s[nt] = acc;
        }

        // Online softmax (row sums ride in o[8] via the ones columns of V).
#pragma unroll
        for (int v = 0; v < 8; ++v) {
            float t = fmaxf(fmaxf(s[0][v], s[1][v]), fmaxf(s[2][v], s[3][v]));
            t = half_reduce_max(t);
            const float nm = fmaxf(m[v], t);
            const float sc = __expf(m[v] - nm);
            m[v] = nm;
#pragma unroll
            for (int dt = 0; dt < 9; ++dt) o[dt][v] *= sc;
#pragma unroll
            for (int nt = 0; nt < 4; ++nt)
                p_lds[wave][v + 8 * half][nt * 16 + l16] = (__bf16)__expf(s[nt][v] - nm);
        }
        wave_lds_fence();

        // O += P @ V' over two 32-key halves, 9 dim-tiles (incl. row-sum tile).
        const v16bf ap0 = load_afrag_bf16(&p_lds[wave][l16][0], 8 * half);
        const v16bf ap1 = load_afrag_bf16(&p_lds[wave][l16][0], 32 + 8 * half);
#pragma unroll
        for (int dt = 0; dt < 9; ++dt) {
            const v16bf b0 = load_bfrag16(&Vs[buf][8 * half][dt * 16 + l16], DVL);
            const v16bf b1 = load_bfrag16(&Vs[buf][32 + 8 * half][dt * 16 + l16], DVL);
            o[dt] = wmma_bf16(ap0, b0, o[dt]);
            o[dt] = wmma_bf16(ap1, b1, o[dt]);
        }
    }

    // Normalize by the accumulated row sums and stage O for the projection.
#pragma unroll
    for (int v = 0; v < 8; ++v) {
        const float rl = 1.0f / o[8][v];
#pragma unroll
        for (int dt = 0; dt < 8; ++dt)
            o_lds[wave][v + 8 * half][dt * 16 + l16] = (__bf16)(o[dt][v] * rl);
    }
    wave_lds_fence();

    // out = relu(O @ Wg^T + bg)
    v16bf ao[4];
#pragma unroll
    for (int kt = 0; kt < 4; ++kt)
        ao[kt] = load_afrag_bf16(&o_lds[wave][l16][0], 32 * kt + 8 * half);

#pragma unroll
    for (int et = 0; et < 8; ++et) {
        const int e = et * 16 + l16;
        v16bf bf[4];
#pragma unroll
        for (int kt = 0; kt < 4; ++kt)
            bf[kt] = load_bfrag16(&wt_g[32 * kt + 8 * half][e], DD);
        v8f acc = {};
#pragma unroll
        for (int kt = 0; kt < 4; ++kt) acc = wmma_bf16(ao[kt], bf[kt], acc);
        const float bias = bg[e];
#pragma unroll
        for (int v = 0; v < 8; ++v) {
            const float r = acc[v] + bias;
            out[(nBase + qr0 + v + 8 * half) * DD + e] = r > 0.0f ? r : 0.0f;
        }
    }
}

// ---------------------------------------------------------------------------
// Launch: ws layout = q(4MB) | kT(4MB) | V(4.5MB padded), all bf16.
// ---------------------------------------------------------------------------
extern "C" void kernel_launch(void* const* d_in, const int* in_sizes, int n_in,
                              void* d_out, int out_size, void* d_ws, size_t ws_size,
                              hipStream_t stream) {
    const float* X  = (const float*)d_in[0];
    const float* Wq = (const float*)d_in[1];
    const float* bq = (const float*)d_in[2];
    const float* Wk = (const float*)d_in[3];
    const float* bk = (const float*)d_in[4];
    const float* Wg = (const float*)d_in[5];
    const float* bg = (const float*)d_in[6];
    float* out = (float*)d_out;

    char* ws = (char*)d_ws;
    unsigned short* q_ws  = (unsigned short*)(ws);
    unsigned short* kT_ws = (unsigned short*)(ws + (size_t)(4 << 20));
    unsigned short* v_ws  = (unsigned short*)(ws + (size_t)(8 << 20));

    qkv_proj_kernel<<<(BB * NN) / 128, 256, 0, stream>>>(X, Wq, bq, Wk, bk,
                                                         q_ws, kT_ws, v_ws);
    attn_out_kernel<<<BB * (NN / 128), 256, 0, stream>>>(q_ws, kT_ws, v_ws,
                                                         Wg, bg, out);
}
